// DeformableConvLayer_20478404067793
// MI455X (gfx1250) — matrix operand order repaired and checked by reading, not compiled, not run on hardware
//
#include <hip/hip_runtime.h>

// MI455X / gfx1250, wave32. Implicit-GEMM 3x3 conv (deformable conv with zero
// offsets == dense conv) using V_WMMA_F32_16X16X32_F16 with f32 accumulate.
//
// GEMM: D[64 outc x 128 cols] = A[64 x 192] * B[192 x 128cols] + bias
//   K = tap*18 + cin  (9 taps x 18 cin = 162, zero-padded to 192 = 6 * 32)
// Block = 256 threads (8 waves); each wave: 16 columns x all 64 out-channels
// (4 accumulator tiles of 16x16 f32 = 32 VGPRs).
// LDS: sA 25.6KB + sB 51.2KB + bias = ~77KB (compile-verified OK on gfx1250;
// 4 blocks/WGP = 32 waves at 308KB of the 320KB LDS pool).
// Output stores are non-temporal: 537MB of the 690MB HBM traffic is
// write-once data that must not evict the L2-resident input rows / weights.

typedef __attribute__((ext_vector_type(16))) _Float16 v16h;
typedef __attribute__((ext_vector_type(8)))  _Float16 v8h;
typedef __attribute__((ext_vector_type(8)))  float    v8f;

#define CIN   18
#define COUT  64
#define HH    256
#define WW    256
#define NN    32
#define KTOT  162      // 9 taps * 18 cin
#define KPAD  192      // 6 * 32 (K per wmma = 32)
#define KP    200      // LDS row pitch in f16: 400B -> bank-conflict-free b128

#define COLS  128      // output columns per block (power of two: shift/mask)
#define BLOCK 256      // 8 waves

__global__ __launch_bounds__(BLOCK)
void deform_conv3x3_wmma_f16(const float* __restrict__ x,
                             const float* __restrict__ w,
                             const float* __restrict__ bias,
                             float* __restrict__ out)
{
    __shared__ _Float16 sA[COUT * KP];   // repacked weights  [outc][k]
    __shared__ _Float16 sB[COLS * KP];   // im2col panel      [col][k]
    __shared__ float    sBias[COUT];

    const int tid   = threadIdx.x;
    const int wbase = blockIdx.x * COLS;   // first output column
    const int h     = blockIdx.y;          // output row
    const int n     = blockIdx.z;          // batch index

    if (tid < COUT) sBias[tid] = bias[tid];

    // ---- stage weights -> sA (f32 -> f16); k = tap*18 + cin, zero-pad k>=162
    // m fastest across threads (weights are tiny and L1/L2-resident).
    for (int i = tid; i < COUT * KPAD; i += BLOCK) {
        const int m = i & (COUT - 1);
        const int k = i >> 6;            // COUT == 64
        float v = 0.f;
        if (k < KTOT) {
            const int tap = k / CIN, cin = k % CIN;
            const int kh = tap / 3, kw = tap % 3;
            v = w[((m * CIN + cin) * 3 + kh) * 3 + kw];
        }
        sA[m * KP + k] = (_Float16)v;
    }

    // ---- stage im2col panel -> sB; column fastest across threads so the
    // global x reads are contiguous along W (coalesced), one k-row at a time.
    for (int i = tid; i < COLS * KPAD; i += BLOCK) {
        const int c = i & (COLS - 1);
        const int k = i >> 7;            // COLS == 128
        float v = 0.f;
        if (k < KTOT) {
            const int tap = k / CIN, cin = k % CIN;
            const int kh = tap / 3, kw = tap % 3;
            const int hh = h + kh - 1;
            const int wc = wbase + c + kw - 1;
            if (hh >= 0 && hh < HH && wc >= 0 && wc < WW)
                v = x[((n * CIN + cin) * HH + hh) * WW + wc];
        }
        sB[c * KP + k] = (_Float16)v;
    }

    __syncthreads();

    const int lane = tid & 31;
    const int wave = tid >> 5;
    const int lm   = lane & 15;   // column / M within tile
    const int lh   = lane >> 4;   // lane-half (K / M split per ISA layouts)
    const int c0   = wave * 16;   // this wave's first column

    // C/D layout: VGPR j holds M=j (lanes 0-15) and M=j+8 (lanes 16-31).
    // Fold bias into the accumulator init.
    v8f acc[4];
    #pragma unroll
    for (int mt = 0; mt < 4; ++mt) {
        #pragma unroll
        for (int j = 0; j < 8; ++j)
            acc[mt][j] = sBias[mt * 16 + j + 8 * lh];
    }

    #pragma unroll
    for (int ks = 0; ks < KPAD / 32; ++ks) {
        const int kb = ks * 32;

        // B (32x16 f16): lanes 0-15 hold K=kb..kb+15 for column lm,
        //                lanes 16-31 hold K=kb+16..kb+31.  16B-aligned b128s.
        const _Float16* bp = &sB[(c0 + lm) * KP + kb + 16 * lh];
        const v8h blo = *(const v8h*)(bp);
        const v8h bhi = *(const v8h*)(bp + 8);
        v16h bmat;
        #pragma unroll
        for (int i = 0; i < 8; ++i) { bmat[i] = blo[i]; bmat[i + 8] = bhi[i]; }

        #pragma unroll
        for (int mt = 0; mt < 4; ++mt) {
            // A (16x32 f16): lane holds K=kb+16*lh..+7 (vgpr0-3) and
            //                K=kb+16*lh+8..+15 (vgpr4-7) for row M=lm.
            const _Float16* ap = &sA[(mt * 16 + lm) * KP + kb + 16 * lh];
            const v8h alo = *(const v8h*)(ap);
            const v8h ahi = *(const v8h*)(ap + 8);
            v16h amat;
            #pragma unroll
            for (int i = 0; i < 8; ++i) { amat[i] = alo[i]; amat[i + 8] = ahi[i]; }

            acc[mt] = __builtin_amdgcn_wmma_f32_16x16x32_f16(
                /*neg_a=*/false, amat, /*neg_b=*/false, bmat,
                /*c_mod=*/(short)0, acc[mt],
                /*reuse_a=*/false, /*reuse_b=*/false);
        }
    }

    // ---- store: element j of tile mt -> outc = mt*16 + j + 8*lh, col = lm
    // Non-temporal: output is write-once streaming data; keep it out of the
    // way of the L2-resident input rows and weights.
    const int col = wbase + c0 + lm;
    #pragma unroll
    for (int mt = 0; mt < 4; ++mt) {
        #pragma unroll
        for (int j = 0; j < 8; ++j) {
            const int oc = mt * 16 + j + 8 * lh;
            __builtin_nontemporal_store(
                acc[mt][j], &out[((n * COUT + oc) * HH + h) * WW + col]);
        }
    }
}

extern "C" void kernel_launch(void* const* d_in, const int* in_sizes, int n_in,
                              void* d_out, int out_size, void* d_ws, size_t ws_size,
                              hipStream_t stream) {
    (void)in_sizes; (void)n_in; (void)d_ws; (void)ws_size; (void)out_size;
    const float* x    = (const float*)d_in[0];
    const float* wgt  = (const float*)d_in[1];
    const float* bias = (const float*)d_in[2];
    float* out = (float*)d_out;

    dim3 grid(WW / COLS, HH, NN);   // (2, 256, 32)
    deform_conv3x3_wmma_f16<<<grid, dim3(BLOCK), 0, stream>>>(x, wgt, bias, out);
}
